// GRU_1503238554285
// MI455X (gfx1250) — matrix operand order repaired and checked by reading, not compiled, not run on hardware
//
#include <hip/hip_runtime.h>

typedef __attribute__((ext_vector_type(16))) __bf16 v16bf;
typedef __attribute__((ext_vector_type(8)))  float  v8f;
typedef __attribute__((ext_vector_type(4)))  float  f32x4;
typedef __attribute__((ext_vector_type(4)))  unsigned int u32x4;

#define B_SZ 128
#define T_SZ 512
#define H_SZ 256
#define D_SZ 256
#define NC   768            // 3*H
#define NROW 65536          // B*T
#define NWG_REC 16

// ---------------- workspace layout (bytes) ----------------
static constexpr size_t OFF_CTR  = 0;                         // grid-barrier counter
static constexpr size_t OFF_WIB  = 256;
static constexpr size_t SZ_W     = (size_t)NC * D_SZ * 2;     // 393216 B per packed bf16 weight
static constexpr size_t OFF_WIA  = OFF_WIB + SZ_W;            // |Wi|
static constexpr size_t OFF_WHB  = OFF_WIA + SZ_W;            // Wh
static constexpr size_t OFF_WHA  = OFF_WHB + SZ_W;            // |Wh|
static constexpr size_t OFF_HA   = OFF_WHA + SZ_W;            // packed h A-tiles, double buffered
static constexpr size_t SZ_HABUF = (size_t)3 * 64 * 32 * 16 * 2; // 3 variants * 64 tiles * 32 lanes * 16 bf16
static constexpr size_t OFF_I    = OFF_HA + 2 * SZ_HABUF;     // i_out (z, lb, ub), fp32
static constexpr size_t SZ_I     = (size_t)NROW * NC * 4;     // per component
static constexpr size_t OUTC     = (size_t)B_SZ * T_SZ * H_SZ; // elements per output component

// ---------------- helpers ----------------
__device__ __forceinline__ unsigned short bfbits(float x) {
  unsigned u = __builtin_bit_cast(unsigned, x);
  unsigned r = (u + 0x7FFFu + ((u >> 16) & 1u)) >> 16;   // round-to-nearest-even
  return (unsigned short)r;
}
__device__ __forceinline__ v8f wmma_bf16(v16bf a, v16bf b, v8f c) {
  return __builtin_amdgcn_wmma_f32_16x16x32_bf16(false, a, false, b, (short)0, c, false, false);
}
__device__ __forceinline__ float sigf(float x) { return 1.f / (1.f + __expf(-x)); }
__device__ __forceinline__ float tanhfast(float x) {
  x = fminf(fmaxf(x, -15.f), 15.f);
  float e = __expf(2.f * x);
  return (e - 1.f) / (e + 1.f);
}
__device__ __forceinline__ void imul(float av, float al, float au,
                                     float bv, float bl, float bu,
                                     float& cv, float& cl, float& cu) {
  float ll = al * bl, lu = al * bu, ul = au * bl, uu = au * bu;
  cl = fminf(fminf(ll, lu), fminf(ul, uu));
  cu = fmaxf(fmaxf(ll, lu), fmaxf(ul, uu));
  cv = av * bv;
}
__device__ __forceinline__ u32x4 pack8(const unsigned short* p) {
  u32x4 v;
  v.x = (unsigned)p[0] | ((unsigned)p[1] << 16);
  v.y = (unsigned)p[2] | ((unsigned)p[3] << 16);
  v.z = (unsigned)p[4] | ((unsigned)p[5] << 16);
  v.w = (unsigned)p[6] | ((unsigned)p[7] << 16);
  return v;
}
__device__ __forceinline__ void ldnt4(float* dst, const float* src) {
  *(f32x4*)dst = __builtin_nontemporal_load((const f32x4*)src);
}

// ============================================================
// Kernel 0a: pack Wi, |Wi|, Wh, |Wh| into bf16 WMMA B-tile lane order.
// B-tile (K=32 x N=16): lane L holds col n=L&15; element i -> K=ks*32 +
// (i<8 ? i : i+8) + (L>=16 ? 8 : 0). Packed: ((ct*8+ks)*32+L)*16.
// ============================================================
__global__ void pack_weights_kernel(const float* __restrict__ Wi,
                                    const float* __restrict__ Wh,
                                    char* __restrict__ ws) {
  int tid = blockIdx.x * 256 + threadIdx.x;   // 0..49151
  int arr = tid / 12288;                      // 0:Wi 1:|Wi| 2:Wh 3:|Wh|
  int rem = tid % 12288;
  int tile = rem >> 5;                        // ct*8+ks, 0..383
  int L = rem & 31;
  int ct = tile >> 3, ks = tile & 7;
  int j = ct * 16 + (L & 15);
  int kb = ks * 32 + ((L >= 16) ? 8 : 0);
  const float* W = (arr < 2) ? Wi : Wh;
  bool isabs = (arr & 1) != 0;
  unsigned short pk[16];
#pragma unroll
  for (int i = 0; i < 16; i++) {
    float v = W[(size_t)j * D_SZ + kb + ((i < 8) ? i : i + 8)];
    if (isabs) v = fabsf(v);
    pk[i] = bfbits(v);
  }
  char* dst = ws + OFF_WIB + (size_t)arr * SZ_W + ((size_t)(tile * 32 + L) * 16) * 2;
  ((u32x4*)dst)[0] = pack8(pk);
  ((u32x4*)dst)[1] = pack8(pk + 8);
}

// ============================================================
// Kernel 0b: pack h0 into A-tile bf16 buffer 0 (val, mid=h0, rad=0).
// A-tile (M=16 x K=32): lane L: m = L&15; element i -> k = ks*32 +
// (i<8 ? i : i+8) + (L>=16 ? 8 : 0).
// ============================================================
__global__ void pack_h0_kernel(const float* __restrict__ h0, char* __restrict__ ws) {
  int tid = blockIdx.x * 256 + threadIdx.x;   // 0..6143
  int var = tid / 2048;
  int rem = tid % 2048;
  int tile = rem >> 5;                        // rt*8+ks, 0..63
  int L = rem & 31;
  int rt = tile >> 3, ks = tile & 7;
  int b = rt * 16 + (L & 15);
  unsigned short pk[16];
#pragma unroll
  for (int i = 0; i < 16; i++) {
    int k = ks * 32 + ((i < 8) ? i : i + 8) + ((L >= 16) ? 8 : 0);
    float v = (var == 2) ? 0.f : h0[(size_t)b * H_SZ + k];
    pk[i] = bfbits(v);
  }
  char* dst = ws + OFF_HA + ((size_t)((var * 64 + tile) * 32 + L) * 16) * 2;
  ((u32x4*)dst)[0] = pack8(pk);
  ((u32x4*)dst)[1] = pack8(pk + 8);
}

// ============================================================
// Kernel 1: time-parallel input-side IBP linear.
// One block per 16-row tile of (B*T). Block packs val/mid/rad A-tiles
// for all 8 k-steps into LDS once (x read exactly once, NT loads),
// then 8 waves each cover 6 of the 48 column tiles x 3 variants.
// Outputs i_z/i_lb/i_ub (NROW x 768) fp32 with NT stores.
// ============================================================
__global__ void __launch_bounds__(256, 1)
iout_gemm_kernel(const float* __restrict__ xv,
                 const float* __restrict__ xl,
                 const float* __restrict__ xu,
                 const float* __restrict__ bi,
                 char* __restrict__ ws) {
  const int w = threadIdx.x >> 5;     // wave 0..7 -> col tiles [w*6, w*6+6)
  const int L = threadIdx.x & 31;
  const int nlane = L & 15;
  const int hi = (L >= 16) ? 1 : 0;
  const int rtbase = blockIdx.x * 16;  // 16 rows of B*T

  __shared__ __align__(32) unsigned short ldsA[3][8][32][16];  // 24 KB

  // ---- cooperative A-tile pack: each thread owns one (ks, lane) slot ----
  {
    int ksP = threadIdx.x >> 5;        // 0..7
    int LP  = threadIdx.x & 31;
    int rowP = rtbase + (LP & 15);
    size_t xo = (size_t)rowP * D_SZ + ksP * 32 + ((LP >= 16) ? 8 : 0);
    float a[16], l_[16], u_[16];
    ldnt4(a + 0,  xv + xo);      ldnt4(a + 4,  xv + xo + 4);
    ldnt4(a + 8,  xv + xo + 16); ldnt4(a + 12, xv + xo + 20);
    ldnt4(l_ + 0,  xl + xo);      ldnt4(l_ + 4,  xl + xo + 4);
    ldnt4(l_ + 8,  xl + xo + 16); ldnt4(l_ + 12, xl + xo + 20);
    ldnt4(u_ + 0,  xu + xo);      ldnt4(u_ + 4,  xu + xo + 4);
    ldnt4(u_ + 8,  xu + xo + 16); ldnt4(u_ + 12, xu + xo + 20);
    unsigned short pv[16], pm[16], pr[16];
#pragma unroll
    for (int i = 0; i < 16; i++) {
      pv[i] = bfbits(a[i]);
      pm[i] = bfbits((l_[i] + u_[i]) * 0.5f);
      pr[i] = bfbits((u_[i] - l_[i]) * 0.5f);
    }
    *(u32x4*)&ldsA[0][ksP][LP][0] = pack8(pv);
    *(u32x4*)&ldsA[0][ksP][LP][8] = pack8(pv + 8);
    *(u32x4*)&ldsA[1][ksP][LP][0] = pack8(pm);
    *(u32x4*)&ldsA[1][ksP][LP][8] = pack8(pm + 8);
    *(u32x4*)&ldsA[2][ksP][LP][0] = pack8(pr);
    *(u32x4*)&ldsA[2][ksP][LP][8] = pack8(pr + 8);
  }
  __syncthreads();

  const v16bf* WiBp = (const v16bf*)(ws + OFF_WIB);
  const v16bf* WiAp = (const v16bf*)(ws + OFF_WIA);
  float* iZ  = (float*)(ws + OFF_I);
  float* iLB = (float*)(ws + OFF_I + SZ_I);
  float* iUB = (float*)(ws + OFF_I + 2 * SZ_I);

  float biv[6];
#pragma unroll
  for (int c = 0; c < 6; c++) biv[c] = bi[(w * 6 + c) * 16 + nlane];

  v8f accz[6], accm[6], accr[6];
#pragma unroll
  for (int c = 0; c < 6; c++) { accz[c] = (v8f)0.f; accm[c] = (v8f)0.f; accr[c] = (v8f)0.f; }

#pragma unroll
  for (int ks = 0; ks < 8; ks++) {
    v16bf Av = *(const v16bf*)&ldsA[0][ks][L][0];
    v16bf Am = *(const v16bf*)&ldsA[1][ks][L][0];
    v16bf Ar = *(const v16bf*)&ldsA[2][ks][L][0];
#pragma unroll
    for (int c = 0; c < 6; c++) {
      int ct = w * 6 + c;
      int bidx = (ct * 8 + ks) * 32 + L;
      v16bf Bw = WiBp[bidx];
      v16bf Ba = WiAp[bidx];
      accz[c] = wmma_bf16(Av, Bw, accz[c]);
      accm[c] = wmma_bf16(Am, Bw, accm[c]);
      accr[c] = wmma_bf16(Ar, Ba, accr[c]);
    }
  }

#pragma unroll
  for (int c = 0; c < 6; c++) {
    int col = (w * 6 + c) * 16 + nlane;
#pragma unroll
    for (int r = 0; r < 8; r++) {
      size_t n = (size_t)(rtbase + r + hi * 8);
      size_t o = n * NC + col;
      float z  = accz[c][r] + biv[c];
      float mu = accm[c][r] + biv[c];
      float rr = accr[c][r];
      __builtin_nontemporal_store(z,       iZ  + o);
      __builtin_nontemporal_store(mu - rr, iLB + o);
      __builtin_nontemporal_store(mu + rr, iUB + o);
    }
  }
}

// ============================================================
// Kernel 2: persistent recurrent IBP-GRU. 16 WGs x 256 threads.
// WG g owns H-columns [g*16, g*16+16). Wh/|Wh| B-tiles for this WG
// (48 KB) are staged in LDS once. Per step: 3 interval GEMMs
// (bf16 WMMA, K=256) -> gates -> h_new -> repack bf16 A-tiles for
// next step (double buffered) + fp32 h history into d_out.
// Cross-WG step sync: monotonic-counter device barrier.
// ============================================================
__global__ void __launch_bounds__(256, 1)
recurrent_kernel(const float* __restrict__ h0,
                 const float* __restrict__ bh,
                 char* __restrict__ ws,
                 float* __restrict__ outp) {
  const int g = blockIdx.x;
  const int w = threadIdx.x >> 5;     // wave = row tile (0..7)
  const int L = threadIdx.x & 31;
  const int nlane = L & 15;
  const int hi = (L >= 16) ? 1 : 0;

  const float* iZ  = (const float*)(ws + OFF_I);
  const float* iLB = (const float*)(ws + OFF_I + SZ_I);
  const float* iUB = (const float*)(ws + OFF_I + 2 * SZ_I);
  unsigned* ctr = (unsigned*)(ws + OFF_CTR);

  __shared__ __align__(32) unsigned short ldsW[2][3][8][32][16]; // 48 KB: [arr][gg][ks][L][16]
  __shared__ float lds_h[3][B_SZ][16];                           // 24 KB: [comp][b][jj]

  // ---- stage this WG's Wh / |Wh| B-tiles into LDS (once) ----
  {
    const v16bf* WhBp = (const v16bf*)(ws + OFF_WHB);
    const v16bf* WhAp = (const v16bf*)(ws + OFF_WHA);
    for (int s = threadIdx.x; s < 2 * 3 * 8 * 32; s += 256) {
      int arr = s >> 9;               // 0..2 here; recomputed below
      int rem = s & 511;
      int gg = rem >> 8 == 0 ? (rem >> 8) : (rem >> 8);  // placeholder, recompute below
      gg = rem / 256;                 // 0..2?  (rem < 512 -> gg in 0..1)  -- use full decode:
      // full decode: s = ((arr*3 + gg)*8 + ks)*32 + l
      int tmp = s;
      int l = tmp & 31; tmp >>= 5;
      int ks = tmp & 7; tmp >>= 3;
      gg = tmp % 3; arr = tmp / 3;
      int ct = gg * 16 + g;
      int bidx = (ct * 8 + ks) * 32 + l;
      v16bf t = (arr == 0) ? WhBp[bidx] : WhAp[bidx];
      *(v16bf*)&ldsW[arr][gg][ks][l][0] = t;
    }
  }
  __syncthreads();

  float bhv[3];
#pragma unroll
  for (int gg = 0; gg < 3; gg++) bhv[gg] = bh[gg * H_SZ + g * 16 + nlane];

  for (int t = 0; t < T_SZ; t++) {
    const v16bf* hAbuf = (const v16bf*)(ws + OFF_HA + (size_t)(t & 1) * SZ_HABUF);

    // ---- 3 interval GEMMs over K=256 ----
    v8f az[3], am[3], ar[3];
#pragma unroll
    for (int gg = 0; gg < 3; gg++) { az[gg] = (v8f)0.f; am[gg] = (v8f)0.f; ar[gg] = (v8f)0.f; }

#pragma unroll
    for (int ks = 0; ks < 8; ks++) {
      int ti = w * 8 + ks;
      v16bf Av  = hAbuf[(0 * 64 + ti) * 32 + L];
      v16bf Amd = hAbuf[(1 * 64 + ti) * 32 + L];
      v16bf Ard = hAbuf[(2 * 64 + ti) * 32 + L];
#pragma unroll
      for (int gg = 0; gg < 3; gg++) {
        v16bf Bw = *(const v16bf*)&ldsW[0][gg][ks][L][0];
        v16bf Ba = *(const v16bf*)&ldsW[1][gg][ks][L][0];
        az[gg] = wmma_bf16(Av,  Bw, az[gg]);
        am[gg] = wmma_bf16(Amd, Bw, am[gg]);
        ar[gg] = wmma_bf16(Ard, Ba, ar[gg]);
      }
    }

    // ---- elementwise interval GRU cell ----
#pragma unroll
    for (int r = 0; r < 8; r++) {
      int b = w * 16 + r + hi * 8;
      size_t nrow = (size_t)b * T_SZ + t;
      float iz[3], il[3], iu[3], hv[3], hl[3], hu[3];
#pragma unroll
      for (int gg = 0; gg < 3; gg++) {
        size_t io = nrow * NC + gg * H_SZ + g * 16 + nlane;
        iz[gg] = iZ[io]; il[gg] = iLB[io]; iu[gg] = iUB[io];
        if (t < T_SZ - 1) __builtin_prefetch(iZ + io + NC, 0, 0);
        float zz = az[gg][r] + bhv[gg];
        float mm = am[gg][r] + bhv[gg];
        float rr = ar[gg][r];
        hv[gg] = zz; hl[gg] = mm - rr; hu[gg] = mm + rr;
      }
      // gates (sigmoid at interval endpoints)
      float rv = sigf(iz[0] + hv[0]), rl = sigf(il[0] + hl[0]), ru = sigf(iu[0] + hu[0]);
      float zv = sigf(iz[1] + hv[1]), zl = sigf(il[1] + hl[1]), zu = sigf(iu[1] + hu[1]);
      // n = tanh(i_state + r (x) h_state)
      float pv, pl, pu;
      imul(rv, rl, ru, hv[2], hl[2], hu[2], pv, pl, pu);
      float nv = tanhfast(iz[2] + pv), nl = tanhfast(il[2] + pl), nu = tanhfast(iu[2] + pu);
      // previous h interval (degenerate h0 at t==0, else our own t-1 output)
      int j = g * 16 + nlane;
      float hpv, hpl, hpu;
      if (t == 0) {
        float h00 = h0[(size_t)b * H_SZ + j];
        hpv = hpl = hpu = h00;
      } else {
        size_t ob = (size_t)b * T_SZ * H_SZ + (size_t)(t - 1) * H_SZ + j;
        hpv = outp[ob]; hpl = outp[OUTC + ob]; hpu = outp[2 * OUTC + ob];
      }
      // h_new = (1-z) (x) n + z (x) h   (interval 1-z flips endpoints)
      float qv, ql, qu, sv, sl, su;
      imul(1.f - zv, 1.f - zu, 1.f - zl, nv, nl, nu, qv, ql, qu);
      imul(zv, zl, zu, hpv, hpl, hpu, sv, sl, su);
      lds_h[0][b][nlane] = qv + sv;
      lds_h[1][b][nlane] = ql + sl;
      lds_h[2][b][nlane] = qu + su;
    }
    __syncthreads();

    // ---- repack h_new slice into next step's bf16 A-tiles ----
    {
      int rt2 = threadIdx.x >> 5;
      int L2 = threadIdx.x & 31;
      int ks2 = g >> 1;
      int half = g & 1;
      int jb = (L2 >= 16) ? 8 : 0;
      int b2 = rt2 * 16 + (L2 & 15);
      unsigned short pkv[8], pkm[8], pkr[8];
#pragma unroll
      for (int i = 0; i < 8; i++) {
        int jj = jb + i;
        float vv = lds_h[0][b2][jj];
        float ll = lds_h[1][b2][jj];
        float uu = lds_h[2][b2][jj];
        pkv[i] = bfbits(vv);
        pkm[i] = bfbits((ll + uu) * 0.5f);
        pkr[i] = bfbits((uu - ll) * 0.5f);
      }
      char* dbuf = ws + OFF_HA + (size_t)((t + 1) & 1) * SZ_HABUF;
      size_t tbase = (size_t)(rt2 * 8 + ks2) * 32 + L2;
      *(u32x4*)(dbuf + ((0 * 2048 + tbase) * 16 + half * 8) * 2) = pack8(pkv);
      *(u32x4*)(dbuf + ((1 * 2048 + tbase) * 16 + half * 8) * 2) = pack8(pkm);
      *(u32x4*)(dbuf + ((2 * 2048 + tbase) * 16 + half * 8) * 2) = pack8(pkr);
    }

    // ---- fp32 h history -> d_out (3,B,T,H) ----
    for (int idx = threadIdx.x; idx < B_SZ * 16; idx += 256) {
      int b3 = idx >> 4;
      int jj = idx & 15;
      size_t ob = (size_t)b3 * T_SZ * H_SZ + (size_t)t * H_SZ + g * 16 + jj;
      outp[ob]            = lds_h[0][b3][jj];
      outp[OUTC + ob]     = lds_h[1][b3][jj];
      outp[2 * OUTC + ob] = lds_h[2][b3][jj];
    }

    // ---- device-wide step barrier (monotonic counter) ----
    __threadfence();
    __syncthreads();
    if (threadIdx.x == 0) {
      atomicAdd(ctr, 1u);
      unsigned target = (unsigned)NWG_REC * (unsigned)(t + 1);
      while (__hip_atomic_load(ctr, __ATOMIC_RELAXED, __HIP_MEMORY_SCOPE_AGENT) < target)
        __builtin_amdgcn_s_sleep(1);
    }
    __syncthreads();
    __threadfence();
  }
}

// ============================================================
extern "C" void kernel_launch(void* const* d_in, const int* in_sizes, int n_in,
                              void* d_out, int out_size, void* d_ws, size_t ws_size,
                              hipStream_t stream) {
  const float* xv = (const float*)d_in[0];
  const float* xl = (const float*)d_in[1];
  const float* xu = (const float*)d_in[2];
  const float* h0 = (const float*)d_in[3];
  const float* Wi = (const float*)d_in[4];
  const float* bi = (const float*)d_in[5];
  const float* Wh = (const float*)d_in[6];
  const float* bh = (const float*)d_in[7];
  float* outp = (float*)d_out;
  char* ws = (char*)d_ws;
  (void)in_sizes; (void)n_in; (void)out_size; (void)ws_size;

  // zero the grid-barrier counter (ws is poisoned, not re-zeroed between replays)
  hipMemsetAsync(ws, 0, 256, stream);

  pack_weights_kernel<<<192, 256, 0, stream>>>(Wi, Wh, ws);
  pack_h0_kernel<<<24, 256, 0, stream>>>(h0, ws);
  iout_gemm_kernel<<<4096, 256, 0, stream>>>(xv, xl, xu, bi, ws);
  recurrent_kernel<<<NWG_REC, 256, 0, stream>>>(h0, bh, ws, outp);
}
